// MambaCAN_noconv_6803228196993
// MI455X (gfx1250) — compile-verified, matrix operand order checked by reading, not run on hardware
//
#include <hip/hip_runtime.h>
#include <hip/hip_bf16.h>
#include <math.h>

// ---------------- model dims ----------------
#define L_SEQ   100
#define DMODEL  264
#define DINNER  528
#define DSTATE  16
#define DCONV   4
#define DTRANK  17
#define EMBED   256
#define HID     128
#define NCLS    4
#define BATCH   256
#define BL      (BATCH * L_SEQ)          // 25600 rows (multiple of 32)
#define DBCW    (DTRANK + 2 * DSTATE)    // 49
#define K1PAD   288                      // 264 -> 9 K-steps of 32
#define N1      1056                     // real N of in_proj
#define N1PAD   1088                     // -> 17 macro-tiles of 64
#define K2PAD   544                      // 528 -> 17 K-steps of 32
#define N2PAD   320                      // 264 -> 5 macro-tiles of 64
#define LN_EPSF 1e-5f

typedef __bf16 bf16_t;
typedef __bf16 v16bf __attribute__((ext_vector_type(16)));
typedef __bf16 v8bf  __attribute__((ext_vector_type(8)));
typedef float  v8f   __attribute__((ext_vector_type(8)));

// ---------------------------------------------------------------------------
// WMMA bf16 fragment load: 16x32 tile, row-major, K-contiguous.
// CDNA5 16-bit A layout (ISA 7.12.2): lane<16 -> M=lane, K={0..7,16..23};
// lane>=16 -> M=lane-16, K={8..15,24..31}.  Two 16B loads per lane.
// B (32x16) uses the mirrored layout with N in place of M, hence weights are
// staged K-contiguous per output column (Bt[N][Kpad]).
// ---------------------------------------------------------------------------
__device__ __forceinline__ v16bf load_frag16x32(const bf16_t* __restrict__ base,
                                                int stride, int lane) {
  const int r  = lane & 15;
  const int kb = (lane >> 4) << 3;          // 0 or 8
  const v8bf lo = *reinterpret_cast<const v8bf*>(base + (size_t)r * stride + kb);
  const v8bf hi = *reinterpret_cast<const v8bf*>(base + (size_t)r * stride + kb + 16);
  v16bf f;
#pragma unroll
  for (int i = 0; i < 8; ++i) { f[i] = lo[i]; f[i + 8] = hi[i]; }
  return f;
}

// ---------------- weight prep: transpose to bf16 [Npad][Kpad], zero-padded -
__global__ void prep_w1t(const float* __restrict__ w, bf16_t* __restrict__ wt) {
  int idx = blockIdx.x * blockDim.x + threadIdx.x;           // n * K1PAD + k
  if (idx >= N1PAD * K1PAD) return;
  int n = idx / K1PAD, k = idx - n * K1PAD;
  wt[idx] = (n < N1 && k < DMODEL) ? (bf16_t)w[(size_t)k * N1 + n] : (bf16_t)0.f;
}

__global__ void prep_wot(const float* __restrict__ w, bf16_t* __restrict__ wt) {
  int idx = blockIdx.x * blockDim.x + threadIdx.x;           // n * K2PAD + k
  if (idx >= N2PAD * K2PAD) return;
  int n = idx / K2PAD, k = idx - n * K2PAD;
  wt[idx] = (n < DMODEL && k < DINNER) ? (bf16_t)w[(size_t)k * DMODEL + n] : (bf16_t)0.f;
}

// ---------------- embed + LN + relu + concat -> bf16 feat [BL][K1PAD] -------
__global__ __launch_bounds__(EMBED)
void feat_kernel(const float* __restrict__ x, const float* __restrict__ ew,
                 const float* __restrict__ eb, const float* __restrict__ lng,
                 const float* __restrict__ lnb, bf16_t* __restrict__ feat) {
  __shared__ float red[EMBED];
  const int row = blockIdx.x;                 // b * L + l
  const int b = row / L_SEQ, l = row - b * L_SEQ;
  const int j = threadIdx.x;
  const float idv = x[(size_t)b * (L_SEQ * 9) + l * 9];
  float e = idv * ew[j] + eb[j];

  red[j] = e; __syncthreads();
  for (int s = EMBED / 2; s > 0; s >>= 1) {
    if (j < s) red[j] += red[j + s];
    __syncthreads();
  }
  const float mu = red[0] * (1.f / EMBED); __syncthreads();
  const float dev = e - mu;
  red[j] = dev * dev; __syncthreads();
  for (int s = EMBED / 2; s > 0; s >>= 1) {
    if (j < s) red[j] += red[j + s];
    __syncthreads();
  }
  const float var = red[0] * (1.f / EMBED);
  float en = dev * rsqrtf(var + LN_EPSF) * lng[j] + lnb[j];
  en = fmaxf(en, 0.f);

  bf16_t* frow = feat + (size_t)row * K1PAD;
  frow[j] = (bf16_t)en;
  if (j < 8)  frow[EMBED + j] = (bf16_t)x[(size_t)b * (L_SEQ * 9) + l * 9 + 1 + j];
  if (j >= 8 && j < 32) frow[EMBED + j] = (bf16_t)0.f;   // pad cols 264..287
}

// ---------------------------------------------------------------------------
// Register-blocked WMMA GEMM: one wave computes a 32M x 64N macro-tile.
// Per K-step: 2 A-frag + 4 B-frag loads (12 x b128) feed 8 WMMAs
// -> 1.5 loads/WMMA instead of 4 for a naive 16x16 wave tile.
// ---------------------------------------------------------------------------
#define GEMM_BODY(KPAD)                                                        \
  const int lane = threadIdx.x;                                                \
  const int mt = blockIdx.x << 5, nt = blockIdx.y << 6;                        \
  const bf16_t* Ab = A  + (size_t)mt * (KPAD);                                 \
  const bf16_t* Bb = Bt + (size_t)nt * (KPAD);                                 \
  v8f acc[2][4];                                                               \
  _Pragma("unroll") for (int i = 0; i < 2; ++i)                                \
  _Pragma("unroll") for (int j = 0; j < 4; ++j)                                \
    acc[i][j] = (v8f){0.f, 0.f, 0.f, 0.f, 0.f, 0.f, 0.f, 0.f};                 \
  _Pragma("unroll") for (int kt = 0; kt < (KPAD) / 32; ++kt) {                 \
    v16bf a0 = load_frag16x32(Ab + kt * 32, (KPAD), lane);                     \
    v16bf a1 = load_frag16x32(Ab + 16 * (size_t)(KPAD) + kt * 32, (KPAD), lane);\
    if (kt + 1 < (KPAD) / 32) {                                                \
      __builtin_prefetch(Ab + (kt + 1) * 32 + (size_t)(lane & 31) * (KPAD), 0, 1); \
      __builtin_prefetch(Bb + (kt + 1) * 32 + (size_t)(lane & 31) * (KPAD), 0, 1); \
      __builtin_prefetch(Bb + (kt + 1) * 32 + (size_t)(32 + (lane & 31)) * (KPAD), 0, 1); \
    }                                                                          \
    _Pragma("unroll") for (int j = 0; j < 4; ++j) {                            \
      v16bf bj = load_frag16x32(Bb + (size_t)(j * 16) * (KPAD) + kt * 32,      \
                                (KPAD), lane);                                 \
      acc[0][j] = __builtin_amdgcn_wmma_f32_16x16x32_bf16(                     \
          false, a0, false, bj, (short)0, acc[0][j], false, false);            \
      acc[1][j] = __builtin_amdgcn_wmma_f32_16x16x32_bf16(                     \
          false, a1, false, bj, (short)0, acc[1][j], false, false);            \
    }                                                                          \
  }

// in_proj: [BL,288] x [288,1088] -> xm f32 (cols<528) | z bf16 (528..1055)
__global__ __launch_bounds__(32)
void gemm_inproj(const bf16_t* __restrict__ A, const bf16_t* __restrict__ Bt,
                 float* __restrict__ xm, bf16_t* __restrict__ zb) {
  GEMM_BODY(K1PAD)
#pragma unroll
  for (int i = 0; i < 2; ++i)
#pragma unroll
    for (int j = 0; j < 4; ++j) {
      const int col = nt + j * 16 + (lane & 15);
      const int rb  = mt + i * 16 + ((lane >> 4) << 3);
#pragma unroll
      for (int r = 0; r < 8; ++r) {
        const float v = acc[i][j][r];
        const size_t row = (size_t)(rb + r);
        if (col < DINNER)      xm[row * DINNER + col] = v;
        else if (col < N1)     zb[row * DINNER + (col - DINNER)] = (bf16_t)v;
      }
    }
}

// out_proj: [BL,544] x [544,320] -> mo f32 (cols<264)
__global__ __launch_bounds__(32)
void gemm_outproj(const bf16_t* __restrict__ A, const bf16_t* __restrict__ Bt,
                  float* __restrict__ mo) {
  GEMM_BODY(K2PAD)
#pragma unroll
  for (int i = 0; i < 2; ++i)
#pragma unroll
    for (int j = 0; j < 4; ++j) {
      const int col = nt + j * 16 + (lane & 15);
      const int rb  = mt + i * 16 + ((lane >> 4) << 3);
#pragma unroll
      for (int r = 0; r < 8; ++r)
        if (col < DMODEL) mo[(size_t)(rb + r) * DMODEL + col] = acc[i][j][r];
    }
}

// ---------------- depthwise conv(4) + SiLU ---------------------------------
__global__ void conv_silu(const float* __restrict__ xm, const float* __restrict__ cw,
                          const float* __restrict__ cb, float* __restrict__ xc) {
  int gid = blockIdx.x * blockDim.x + threadIdx.x;
  if (gid >= BL * DINNER) return;
  const int row = gid / DINNER, d = gid - row * DINNER;
  const int b = row / L_SEQ, l = row - b * L_SEQ;
  float acc = cb[d];
#pragma unroll
  for (int k = 0; k < DCONV; ++k) {
    const int ls = l - (DCONV - 1) + k;
    if (ls >= 0) acc += xm[((size_t)b * L_SEQ + ls) * DINNER + d] * cw[d * DCONV + k];
  }
  xc[gid] = acc / (1.f + expf(-acc));   // silu
}

// ---------------- x_proj: [528] -> [49] per row ----------------------------
__global__ __launch_bounds__(64)
void xproj_kernel(const float* __restrict__ xc, const float* __restrict__ w,
                  float* __restrict__ dbc) {
  __shared__ float sx[DINNER];
  const int row = blockIdx.x;
  for (int i = threadIdx.x; i < DINNER; i += 64) sx[i] = xc[(size_t)row * DINNER + i];
  __syncthreads();
  if (threadIdx.x < DBCW) {
    float s = 0.f;
    for (int k = 0; k < DINNER; ++k) s += sx[k] * w[(size_t)k * DBCW + threadIdx.x];
    dbc[(size_t)row * DBCW + threadIdx.x] = s;
  }
}

// ---------------- fused dt_proj + softplus + selective scan + gate ---------
__global__ __launch_bounds__(256)
void scan_kernel(const float* __restrict__ dbc, const float* __restrict__ xc,
                 const bf16_t* __restrict__ zb, const float* __restrict__ dtw,
                 const float* __restrict__ dtb, const float* __restrict__ alog,
                 const float* __restrict__ Dp, bf16_t* __restrict__ ybf) {
  const int b = blockIdx.x;
  const int d = blockIdx.y * blockDim.x + threadIdx.x;
  const bool valid = d < DINNER;
  const bool padlane = (d >= DINNER) && (d < K2PAD);
  __shared__ float srow[DBCW];

  float wcol[DTRANK], a[DSTATE], h[DSTATE];
  float bias = 0.f, Dd = 0.f;
#pragma unroll
  for (int n = 0; n < DSTATE; ++n) { h[n] = 0.f; a[n] = 0.f; }
#pragma unroll
  for (int r = 0; r < DTRANK; ++r) wcol[r] = 0.f;
  if (valid) {
    bias = dtb[d]; Dd = Dp[d];
#pragma unroll
    for (int r = 0; r < DTRANK; ++r) wcol[r] = dtw[(size_t)r * DINNER + d];
#pragma unroll
    for (int n = 0; n < DSTATE; ++n) a[n] = -expf(alog[(size_t)d * DSTATE + n]);
  }

  for (int l = 0; l < L_SEQ; ++l) {
    const size_t row = (size_t)b * L_SEQ + l;
    __syncthreads();
    if (threadIdx.x < DBCW) srow[threadIdx.x] = dbc[row * DBCW + threadIdx.x];
    __syncthreads();
    if (valid) {
      float pre = bias;
#pragma unroll
      for (int r = 0; r < DTRANK; ++r) pre += srow[r] * wcol[r];
      const float delta = (pre > 20.f) ? pre : log1pf(expf(pre));
      const float xt = xc[row * DINNER + d];
      const float dx = delta * xt;
      float y = 0.f;
#pragma unroll
      for (int n = 0; n < DSTATE; ++n) {
        h[n] = h[n] * expf(delta * a[n]) + dx * srow[DTRANK + n];
        y += h[n] * srow[DTRANK + DSTATE + n];
      }
      const float zv = (float)zb[row * DINNER + d];
      const float g = zv / (1.f + expf(-zv));        // silu(z)
      ybf[row * K2PAD + d] = (bf16_t)((y + xt * Dd) * g);
    } else if (padlane) {
      ybf[row * K2PAD + d] = (bf16_t)0.f;            // K-pad for out_proj WMMA
    }
  }
}

// ---------------- max-pool over L ------------------------------------------
__global__ void pool_kernel(const float* __restrict__ mo, float* __restrict__ pooled) {
  int idx = blockIdx.x * blockDim.x + threadIdx.x;
  if (idx >= BATCH * DMODEL) return;
  const int b = idx / DMODEL, n = idx - b * DMODEL;
  float m = -3.4e38f;
  for (int l = 0; l < L_SEQ; ++l)
    m = fmaxf(m, mo[((size_t)b * L_SEQ + l) * DMODEL + n]);
  pooled[idx] = m;
}

// ---------------- head: fc1+relu+fc2 ---------------------------------------
__global__ __launch_bounds__(HID)
void head_kernel(const float* __restrict__ pooled, const float* __restrict__ w1,
                 const float* __restrict__ b1, const float* __restrict__ w2,
                 const float* __restrict__ b2, float* __restrict__ out) {
  __shared__ float sp[DMODEL];
  __shared__ float sh[HID];
  const int b = blockIdx.x, t = threadIdx.x;
  for (int i = t; i < DMODEL; i += HID) sp[i] = pooled[(size_t)b * DMODEL + i];
  __syncthreads();
  float acc = b1[t];
  for (int k = 0; k < DMODEL; ++k) acc += sp[k] * w1[(size_t)k * HID + t];
  sh[t] = fmaxf(acc, 0.f);
  __syncthreads();
  if (t < NCLS) {
    float o = b2[t];
    for (int k = 0; k < HID; ++k) o += sh[k] * w2[(size_t)k * NCLS + t];
    out[(size_t)b * NCLS + t] = o;
  }
}

// ---------------------------------------------------------------------------
extern "C" void kernel_launch(void* const* d_in, const int* in_sizes, int n_in,
                              void* d_out, int out_size, void* d_ws, size_t ws_size,
                              hipStream_t stream) {
  (void)in_sizes; (void)n_in; (void)out_size; (void)ws_size;
  const float* x    = (const float*)d_in[0];
  const float* ew   = (const float*)d_in[1];
  const float* ebi  = (const float*)d_in[2];
  const float* lng  = (const float*)d_in[3];
  const float* lnb  = (const float*)d_in[4];
  const float* ipw  = (const float*)d_in[5];
  const float* cw   = (const float*)d_in[6];
  const float* cb   = (const float*)d_in[7];
  const float* xpw  = (const float*)d_in[8];
  const float* dtw  = (const float*)d_in[9];
  const float* dtb  = (const float*)d_in[10];
  const float* alog = (const float*)d_in[11];
  const float* Dp   = (const float*)d_in[12];
  const float* opw  = (const float*)d_in[13];
  const float* w1   = (const float*)d_in[14];
  const float* b1   = (const float*)d_in[15];
  const float* w2   = (const float*)d_in[16];
  const float* b2   = (const float*)d_in[17];

  size_t off = 0;
  auto take = [&](size_t bytes) -> void* {
    void* p = (void*)((char*)d_ws + off);
    off = (off + bytes + 255) & ~(size_t)255;
    return p;
  };
  bf16_t* W1t   = (bf16_t*)take((size_t)N1PAD * K1PAD * 2);    // 0.6 MB
  bf16_t* Wot   = (bf16_t*)take((size_t)N2PAD * K2PAD * 2);    // 0.3 MB
  bf16_t* featb = (bf16_t*)take((size_t)BL * K1PAD * 2);       // 14.8 MB
  float*  xm    = (float*) take((size_t)BL * DINNER * 4);      // 54 MB
  bf16_t* zb    = (bf16_t*)take((size_t)BL * DINNER * 2);      // 27 MB
  float*  xc    = (float*) take((size_t)BL * DINNER * 4);      // 54 MB
  float*  dbc   = (float*) take((size_t)BL * DBCW * 4);        // 5 MB
  bf16_t* ybf   = (bf16_t*)take((size_t)BL * K2PAD * 2);       // 27.9 MB
  float*  mo    = (float*) take((size_t)BL * DMODEL * 4);      // 27 MB
  float*  pool  = (float*) take((size_t)BATCH * DMODEL * 4);   // 0.3 MB
  // ~211 MB workspace; hot working set per stage stays inside the 192MB L2.

  prep_w1t<<<(N1PAD * K1PAD + 255) / 256, 256, 0, stream>>>(ipw, W1t);
  prep_wot<<<(N2PAD * K2PAD + 255) / 256, 256, 0, stream>>>(opw, Wot);
  feat_kernel<<<BL, EMBED, 0, stream>>>(x, ew, ebi, lng, lnb, featb);
  gemm_inproj<<<dim3(BL / 32, N1PAD / 64), 32, 0, stream>>>(featb, W1t, xm, zb);
  conv_silu<<<(BL * DINNER + 255) / 256, 256, 0, stream>>>(xm, cw, cb, xc);
  xproj_kernel<<<BL, 64, 0, stream>>>(xc, xpw, dbc);
  scan_kernel<<<dim3(BATCH, 3), 256, 0, stream>>>(dbc, xc, zb, dtw, dtb, alog, Dp, ybf);
  gemm_outproj<<<dim3(BL / 32, N2PAD / 64), 32, 0, stream>>>(ybf, Wot, mo);
  pool_kernel<<<(BATCH * DMODEL + 255) / 256, 256, 0, stream>>>(mo, pool);
  head_kernel<<<BATCH, HID, 0, stream>>>(pool, w1, b1, w2, b2, (float*)d_out);
}
